// GINModel_52690658787578
// MI455X (gfx1250) — matrix operand order repaired and checked by reading, not compiled, not run on hardware
//
#include <hip/hip_runtime.h>

typedef __bf16 bf16_t;
typedef __attribute__((ext_vector_type(16))) __bf16 v16bf;
typedef __attribute__((ext_vector_type(8)))  __bf16 v8bf;
typedef __attribute__((ext_vector_type(8)))  float  v8f;

// b128 payload type for the async-to-LDS builtin (matches clang's signature:
// pointer-to-<4 x i32>, global AS1 source, LDS AS3 destination)
typedef int v4i_t __attribute__((vector_size(16)));
typedef __attribute__((address_space(1))) v4i_t* gptr_b128;
typedef __attribute__((address_space(3))) v4i_t* lptr_b128;

#define BN_EPS 1e-5f

// ---------------------------------------------------------------------------
// gfx1250 async global->LDS staging (ASYNCcnt path), with sync fallback
// ---------------------------------------------------------------------------
#if __has_builtin(__builtin_amdgcn_global_load_async_to_lds_b128)
#define GIN_ASYNC_LDS 1
#else
#define GIN_ASYNC_LDS 0
#endif

__device__ __forceinline__ void stage_b128(const bf16_t* g, bf16_t* l) {
#if GIN_ASYNC_LDS
    __builtin_amdgcn_global_load_async_to_lds_b128(
        (gptr_b128)(g), (lptr_b128)(l), 0, 0);
#else
    *reinterpret_cast<v8bf*>(l) = *reinterpret_cast<const v8bf*>(g);
#endif
}

__device__ __forceinline__ void wait_async0() {
#if GIN_ASYNC_LDS
#if __has_builtin(__builtin_amdgcn_s_wait_asynccnt)
    __builtin_amdgcn_s_wait_asynccnt(0);
#else
    asm volatile("s_wait_asynccnt 0x0" ::: "memory");
#endif
#endif
}

// ---------------------------------------------------------------------------
// Edge aggregation: h[dst] += x[src]  (h pre-initialized with x via D2D copy)
// grid: (ceil(E/256), D/4), block 256
// ---------------------------------------------------------------------------
__global__ void edge_agg_kernel(const float* __restrict__ x,
                                const long long* __restrict__ ei,
                                float* __restrict__ h, int E, int D) {
    int e = blockIdx.x * blockDim.x + threadIdx.x;
    if (e >= E) return;
    int c = blockIdx.y * 4;
    int s = (int)ei[e];
    int d = (int)ei[(size_t)E + e];
    const float4 v = *(const float4*)(x + (size_t)s * D + c);
    float* p = h + (size_t)d * D + c;
    unsafeAtomicAdd(p + 0, v.x);
    unsafeAtomicAdd(p + 1, v.y);
    unsafeAtomicAdd(p + 2, v.z);
    unsafeAtomicAdd(p + 3, v.w);
}

// ---------------------------------------------------------------------------
// f32 -> bf16 elementwise (n multiple of 4)
// ---------------------------------------------------------------------------
__global__ void cvt_f32_bf16_kernel(const float* __restrict__ in,
                                    unsigned short* __restrict__ out_,
                                    long long n) {
    long long i = ((long long)blockIdx.x * blockDim.x + threadIdx.x) * 4;
    if (i >= n) return;
    bf16_t* out = reinterpret_cast<bf16_t*>(out_);
    float4 v = *(const float4*)(in + i);
    out[i + 0] = (bf16_t)v.x;
    out[i + 1] = (bf16_t)v.y;
    out[i + 2] = (bf16_t)v.z;
    out[i + 3] = (bf16_t)v.w;
}

// ---------------------------------------------------------------------------
// Weight transpose + f32 -> bf16:  Wt[j,i] = W[i,j],  W is [K, Dout]
// ---------------------------------------------------------------------------
__global__ void wtrans_bf16_kernel(const float* __restrict__ W,
                                   unsigned short* __restrict__ Wt_,
                                   int K, int Dout) {
    int j = blockIdx.x * blockDim.x + threadIdx.x;  // Dout
    int i = blockIdx.y * blockDim.y + threadIdx.y;  // K
    if (i >= K || j >= Dout) return;
    bf16_t* Wt = reinterpret_cast<bf16_t*>(Wt_);
    Wt[(size_t)j * K + i] = (bf16_t)W[(size_t)i * Dout + j];
}

// ---------------------------------------------------------------------------
// WMMA bf16 GEMM:  C[N,Dout] = A[N,K] @ W  using Bt = W^T stored [Dout,K].
// Block: 256 thr = 8 waves; tile = 128 rows x 64 cols, K-step 32.
// B tile staged to LDS once per block per K-step (double-buffered, async on
// gfx1250), shared by all 8 waves; A fragments software-pipelined from global.
//
// A fragment layout (16-bit A 16x32, ISA 7.12.2): lane L holds row (L&15);
//   elems 0..7  = A[row, k + 8*(L>>4) .. +7]
//   elems 8..15 = A[row, k + 16 + 8*(L>>4) .. +7]
// B fragment (16-bit B 32x16): lane L holds col (L&15);
//   elems 0..7  = Bt[col, k + 8*(L>>4) .. +7]
//   elems 8..15 = Bt[col, k + 16 + 8*(L>>4) .. +7]
// C/D: lane L holds col (L&15); vgpr r holds row 8*(L>>4) + r.
// ---------------------------------------------------------------------------
template <bool RELU, bool OUT_BF16>
__global__ __launch_bounds__(256) void gemm_wmma_kernel(
        const unsigned short* __restrict__ A_,
        const unsigned short* __restrict__ Bt_,
        const float* __restrict__ bias,
        void* __restrict__ Cout,
        int N, int K, int Dout) {
    const bf16_t* A  = reinterpret_cast<const bf16_t*>(A_);
    const bf16_t* Bt = reinterpret_cast<const bf16_t*>(Bt_);

    __shared__ bf16_t Bs[2][64][32];   // 8 KB, double-buffered B tile

    const int tid  = threadIdx.x;
    const int lane = tid & 31;
    const int wave = tid >> 5;                        // 0..7
    const int col0 = blockIdx.y * 64;
    const int l15  = lane & 15;
    const int hi   = lane >> 4;

    // Row strip for this wave. Clamp loads so OOB strips still execute the
    // full loop (barriers must be reached by every wave); their stores are
    // guarded off below. N is a multiple of 16 here.
    const int row0 = (blockIdx.x * 8 + wave) * 16;
    int row0c = row0;
    if (row0c > N - 16) row0c = N - 16;
    const bf16_t* aptr = A + (size_t)(row0c + l15) * K;

    // Staging coords: each of the 256 threads copies one b128 per K-step.
    const int sc = tid >> 2;          // 0..63 : column within 64-col tile
    const int sk = (tid & 3) * 8;     // 0,8,16,24 : k offset (bf16 units)
    const bf16_t* bstage = Bt + (size_t)(col0 + sc) * K + sk;

    const int KT = K >> 5;

    // prologue: stage K-block 0 into buffer 0
    stage_b128(bstage, &Bs[0][sc][sk]);
    wait_async0();
    __syncthreads();

    v8f acc[4] = {v8f{}, v8f{}, v8f{}, v8f{}};
    v8bf a0 = *(const v8bf*)(aptr + hi * 8);
    v8bf a1 = *(const v8bf*)(aptr + 16 + hi * 8);

    for (int kt = 0; kt < KT; ++kt) {
        const int cur = kt & 1;
        const int k = kt * 32;
        const bool more = (kt + 1 < KT);

        // kick off next K-block: B tile async copy + A fragment prefetch
        if (more)
            stage_b128(bstage + k + 32, &Bs[cur ^ 1][sc][sk]);

        v8bf na0, na1;
        if (more) {
            __builtin_prefetch(aptr + k + 64, 0, 0);
            na0 = *(const v8bf*)(aptr + k + 32 + hi * 8);
            na1 = *(const v8bf*)(aptr + k + 48 + hi * 8);
        }

        const v16bf afrag = __builtin_shufflevector(a0, a1,
            0, 1, 2, 3, 4, 5, 6, 7, 8, 9, 10, 11, 12, 13, 14, 15);

#pragma unroll
        for (int nt = 0; nt < 4; ++nt) {
            const bf16_t* bl = &Bs[cur][nt * 16 + l15][hi * 8];
            const v8bf b0 = *(const v8bf*)(bl);
            const v8bf b1 = *(const v8bf*)(bl + 16);
            const v16bf bfrag = __builtin_shufflevector(b0, b1,
                0, 1, 2, 3, 4, 5, 6, 7, 8, 9, 10, 11, 12, 13, 14, 15);
            acc[nt] = __builtin_amdgcn_wmma_f32_16x16x32_bf16(
                false, afrag, false, bfrag, (short)0, acc[nt], false, false);
        }

        a0 = na0;
        a1 = na1;
        if (more) wait_async0();   // next-stage copy landed
        __syncthreads();           // safe to overwrite / read next buffer
    }

    // epilogue: bias (+ReLU), store bf16 or f32; guard with UNclamped rows
#pragma unroll
    for (int nt = 0; nt < 4; ++nt) {
        const int col = col0 + nt * 16 + l15;
        const float bv = bias[col];
#pragma unroll
        for (int r = 0; r < 8; ++r) {
            const int row = row0 + hi * 8 + r;
            if (row < N) {
                float v = acc[nt][r] + bv;
                if (RELU) v = fmaxf(v, 0.0f);
                if (OUT_BF16)
                    reinterpret_cast<bf16_t*>(Cout)[(size_t)row * Dout + col] =
                        (bf16_t)v;
                else
                    reinterpret_cast<float*>(Cout)[(size_t)row * Dout + col] = v;
            }
        }
    }
}

// ---------------------------------------------------------------------------
// BatchNorm partial column sums (sum, sum of squares) over a row chunk
// grid: (D/128, 64), block 128
// ---------------------------------------------------------------------------
__global__ void bn_partial_kernel(const float* __restrict__ h,
                                  float* __restrict__ sums,
                                  float* __restrict__ sumsq,
                                  int N, int D) {
    int col = blockIdx.x * blockDim.x + threadIdx.x;
    int chunk = (N + 63) / 64;
    int r0 = blockIdx.y * chunk;
    int r1 = min(N, r0 + chunk);
    float s = 0.0f, ss = 0.0f;
    for (int r = r0; r < r1; ++r) {
        float v = h[(size_t)r * D + col];
        s += v;
        ss += v * v;
    }
    unsafeAtomicAdd(&sums[col], s);
    unsafeAtomicAdd(&sumsq[col], ss);
}

__global__ void bn_finalize_kernel(const float* __restrict__ sums,
                                   const float* __restrict__ sumsq,
                                   const float* __restrict__ gamma,
                                   const float* __restrict__ beta,
                                   float* __restrict__ scale,
                                   float* __restrict__ shift,
                                   int N, int D) {
    int c = blockIdx.x * blockDim.x + threadIdx.x;
    if (c >= D) return;
    float invN = 1.0f / (float)N;
    float mu = sums[c] * invN;
    float var = sumsq[c] * invN - mu * mu;   // biased var, as in torch
    float sc = gamma[c] * rsqrtf(var + BN_EPS);
    scale[c] = sc;
    shift[c] = beta[c] - mu * sc;
}

__global__ void bn_apply_kernel(const float* __restrict__ h,
                                const float* __restrict__ scale,
                                const float* __restrict__ shift,
                                float* __restrict__ y,
                                long long n, int Dmask) {
    long long i = ((long long)blockIdx.x * blockDim.x + threadIdx.x) * 4;
    if (i >= n) return;
    float4 v = *(const float4*)(h + i);
    int c = (int)(i & Dmask);
    float4 o;
    o.x = fmaxf(v.x * scale[c + 0] + shift[c + 0], 0.0f);
    o.y = fmaxf(v.y * scale[c + 1] + shift[c + 1], 0.0f);
    o.z = fmaxf(v.z * scale[c + 2] + shift[c + 2], 0.0f);
    o.w = fmaxf(v.w * scale[c + 3] + shift[c + 3], 0.0f);
    *(float4*)(y + i) = o;
}

// ---------------------------------------------------------------------------
// Graph mean pool: per-graph sums + counts via atomics
// grid: (ceil(N/256), D/4), block 256
// ---------------------------------------------------------------------------
__global__ void pool_sum_kernel(const float* __restrict__ y,
                                const long long* __restrict__ batch,
                                float* __restrict__ psum,
                                float* __restrict__ pcnt,
                                int N, int D) {
    int node = blockIdx.x * blockDim.x + threadIdx.x;
    if (node >= N) return;
    int c = blockIdx.y * 4;
    int g = (int)batch[node];
    float4 v = *(const float4*)(y + (size_t)node * D + c);
    float* p = psum + (size_t)g * D + c;
    unsafeAtomicAdd(p + 0, v.x);
    unsafeAtomicAdd(p + 1, v.y);
    unsafeAtomicAdd(p + 2, v.z);
    unsafeAtomicAdd(p + 3, v.w);
    if (c == 0) unsafeAtomicAdd(&pcnt[g], 1.0f);
}

// out[g,o] = relu( mean_pool[g,:] @ wo[:,o] + bo[o] );  grid 64, block 64
__global__ void out_proj_kernel(const float* __restrict__ psum,
                                const float* __restrict__ pcnt,
                                const float* __restrict__ wo,
                                const float* __restrict__ bo,
                                float* __restrict__ out,
                                int D, int O) {
    int g = blockIdx.x;
    int o = threadIdx.x;
    float inv = 1.0f / fmaxf(pcnt[g], 1.0f);
    float acc = 0.0f;
    for (int k = 0; k < D; ++k)
        acc += psum[(size_t)g * D + k] * wo[(size_t)k * O + o];
    out[(size_t)g * O + o] = fmaxf(acc * inv + bo[o], 0.0f);
}

// ---------------------------------------------------------------------------
// Host orchestration
// ---------------------------------------------------------------------------
extern "C" void kernel_launch(void* const* d_in, const int* in_sizes, int n_in,
                              void* d_out, int out_size, void* d_ws, size_t ws_size,
                              hipStream_t stream) {
    const float* x        = (const float*)d_in[0];
    const long long* ei   = (const long long*)d_in[1];
    const long long* batch= (const long long*)d_in[2];
    const float* wo       = (const float*)d_in[21];
    const float* bo       = (const float*)d_in[22];

    const int N = in_sizes[0] / 128;
    const int E = in_sizes[1] / 2;
    const size_t NS = (size_t)N * 512;

    // carve workspace (256B aligned chunks)
    size_t off = 0;
    auto carve = [&](size_t bytes) -> void* {
        void* p = (char*)d_ws + off;
        off += (bytes + 255) & ~(size_t)255;
        return p;
    };
    float* Y0   = (float*)carve(NS * 4);
    float* Y1   = (float*)carve(NS * 4);
    float* HAG  = (float*)carve(NS * 4);            // agg buffer, reused as pre-BN
    unsigned short* ABF = (unsigned short*)carve(NS * 2);
    unsigned short* H1B = (unsigned short*)carve(NS * 2);
    unsigned short* W1T[3], *W2T[3];
    const int dims[4] = {128, 128, 256, 512};
    for (int l = 0; l < 3; ++l) {
        W1T[l] = (unsigned short*)carve((size_t)dims[l] * dims[l + 1] * 2);
        W2T[l] = (unsigned short*)carve((size_t)dims[l + 1] * dims[l + 1] * 2);
    }
    float* SUMS  = (float*)carve(512 * 4);
    float* SUMSQ = (float*)carve(512 * 4);
    float* SCALE = (float*)carve(512 * 4);
    float* SHIFT = (float*)carve(512 * 4);
    float* PSUM  = (float*)carve(64 * 512 * 4);
    float* PCNT  = (float*)carve(64 * 4);

    auto run_layer = [&](const float* xin, int Din, int Dout, int wi,
                         unsigned short* w1t, unsigned short* w2t, float* yout) {
        const float* w1 = (const float*)d_in[wi + 0];
        const float* b1 = (const float*)d_in[wi + 1];
        const float* w2 = (const float*)d_in[wi + 2];
        const float* b2 = (const float*)d_in[wi + 3];
        const float* gn = (const float*)d_in[wi + 4];
        const float* bn = (const float*)d_in[wi + 5];

        // transpose weights to bf16 [Dout,K]
        {
            dim3 b(16, 16);
            dim3 g1((Dout + 15) / 16, (Din + 15) / 16);
            wtrans_bf16_kernel<<<g1, b, 0, stream>>>(w1, w1t, Din, Dout);
            dim3 g2((Dout + 15) / 16, (Dout + 15) / 16);
            wtrans_bf16_kernel<<<g2, b, 0, stream>>>(w2, w2t, Dout, Dout);
        }
        // h = x + sum_{j->i} x_j
        (void)hipMemcpyAsync(HAG, xin, (size_t)N * Din * 4,
                             hipMemcpyDeviceToDevice, stream);
        edge_agg_kernel<<<dim3((E + 255) / 256, Din / 4), 256, 0, stream>>>(
            xin, ei, HAG, E, Din);
        // convert h to bf16
        long long n1 = (long long)N * Din;
        cvt_f32_bf16_kernel<<<(unsigned)((n1 / 4 + 255) / 256), 256, 0, stream>>>(
            HAG, ABF, n1);
        // MLP layer 1: relu(h @ w1 + b1) -> bf16
        gemm_wmma_kernel<true, true><<<dim3((N + 127) / 128, Dout / 64), 256, 0,
                                       stream>>>(ABF, w1t, b1, H1B, N, Din, Dout);
        // MLP layer 2: (. @ w2 + b2) -> f32 pre-BN (reuse HAG)
        gemm_wmma_kernel<false, false><<<dim3((N + 127) / 128, Dout / 64), 256, 0,
                                         stream>>>(H1B, w2t, b2, HAG, N, Dout, Dout);
        // BatchNorm (batch stats) + ReLU
        (void)hipMemsetAsync(SUMS, 0, Dout * 4, stream);
        (void)hipMemsetAsync(SUMSQ, 0, Dout * 4, stream);
        bn_partial_kernel<<<dim3(Dout / 128, 64), 128, 0, stream>>>(
            HAG, SUMS, SUMSQ, N, Dout);
        bn_finalize_kernel<<<(Dout + 127) / 128, 128, 0, stream>>>(
            SUMS, SUMSQ, gn, bn, SCALE, SHIFT, N, Dout);
        long long n2 = (long long)N * Dout;
        bn_apply_kernel<<<(unsigned)((n2 / 4 + 255) / 256), 256, 0, stream>>>(
            HAG, SCALE, SHIFT, yout, n2, Dout - 1);
    };

    run_layer(x,  128, 128,  3, W1T[0], W2T[0], Y0);
    run_layer(Y0, 128, 256,  9, W1T[1], W2T[1], Y1);
    run_layer(Y1, 256, 512, 15, W1T[2], W2T[2], Y0);

    // global mean pool + output projection
    (void)hipMemsetAsync(PSUM, 0, 64 * 512 * 4, stream);
    (void)hipMemsetAsync(PCNT, 0, 64 * 4, stream);
    pool_sum_kernel<<<dim3((N + 255) / 256, 512 / 4), 256, 0, stream>>>(
        Y0, batch, PSUM, PCNT, N, 512);
    out_proj_kernel<<<64, 64, 0, stream>>>(PSUM, PCNT, wo, bo, (float*)d_out,
                                           512, 64);
}
